// SelfAttentionConv_90288802496613
// MI455X (gfx1250) — matrix operand order repaired and checked by reading, not compiled, not run on hardware
//
#include <hip/hip_runtime.h>
#include <hip/hip_bf16.h>

typedef __attribute__((ext_vector_type(16))) _Float16 v16h;
typedef __attribute__((ext_vector_type(8)))  float    v8f;
typedef __attribute__((ext_vector_type(4)))  unsigned int v4u;
typedef _Float16 half_t;

#define BB 4
#define TT 2048
#define TP 2056   // TT + 8 (causal left pad rows)
#define KD 128

union HFrag {
    v16h v;
    _Float16 h[16];
    unsigned int u[8];
    v4u q[2];
};

union Pack8 {
    v4u q;
    _Float16 h[8];
};

__device__ __forceinline__ v8f zero8() {
    return (v8f){0.f, 0.f, 0.f, 0.f, 0.f, 0.f, 0.f, 0.f};
}

// A-matrix (16x32 f16): lane-half h holds K = 8h..8h+7 (VGPR0..3) and 16+8h..23+8h (VGPR4..7)
__device__ __forceinline__ void load_a_frag(HFrag& f, const half_t* p, int half16) {
    f.q[0] = *(const v4u*)(p + 8 * half16);
    f.q[1] = *(const v4u*)(p + 16 + 8 * half16);
}
// B-matrix (32x16 f16): lane-half h holds K = 8h..8h+7 (VGPR0..3) and 16+8h..23+8h (VGPR4..7)
__device__ __forceinline__ void load_b_frag(HFrag& f, const half_t* p, int half16) {
    f.q[0] = *(const v4u*)(p + 8 * half16);
    f.q[1] = *(const v4u*)(p + 16 + 8 * half16);
}

__device__ __forceinline__ float silu_f(float x) {
    return x / (1.0f + __expf(-x));
}

// ---------------- pack kernels ----------------

__global__ void cvt_f32_f16_kernel(const float* __restrict__ src,
                                   half_t* __restrict__ dst, int n) {
    int i = blockIdx.x * blockDim.x + threadIdx.x;
    if (i < n) dst[i] = (half_t)src[i];
}

// x [B,2048,128] -> xpad [B,2056,128] with 8 leading zero rows per batch
__global__ void pack_xpad_kernel(const float* __restrict__ x,
                                 half_t* __restrict__ xpad) {
    int idx = blockIdx.x * blockDim.x + threadIdx.x;
    if (idx >= BB * TP * KD) return;
    int i    = idx & 127;
    int rest = idx >> 7;
    int tau  = rest % TP;
    int b    = rest / TP;
    half_t v = (half_t)0.0f;
    if (tau >= 8) v = (half_t)x[((size_t)b * TT + (tau - 8)) * KD + i];
    xpad[idx] = v;
}

// W: [1024][128][5] (c,i,j) -> A: [1024][640] with kdim = j*128 + i
__global__ void repack_w_kernel(const float* __restrict__ W,
                                half_t* __restrict__ A) {
    int idx = blockIdx.x * blockDim.x + threadIdx.x;
    if (idx >= 1024 * 640) return;
    int c = idx / 640;
    int k = idx - c * 640;
    int j = k >> 7;
    int i = k & 127;
    A[idx] = (half_t)W[(c * 128 + i) * 5 + j];
}

// ---------------- conv Q/K (dilated, causal via pre-padded input) ----------------
// out[b,c,t] = silu( sum_{i,j} xpad[b, t+2j, i] * Aw[c, j*128+i] + bias[c] ) * scale
// scattered to scrambled attention layout Q2[m=b*8+t/256][s=(t%256)*8 + c%8][d=c/8]
__global__ void __launch_bounds__(128) conv_qk_kernel(
    const half_t* __restrict__ xpad,  // [B,2056,128] f16
    const half_t* __restrict__ Aw,    // [1024,640] f16
    const float*  __restrict__ bias,  // [1024]
    half_t* __restrict__ out,         // [32,2048,128] f16 (scrambled)
    float scale) {
    const int lane   = threadIdx.x & 31;
    const int w      = threadIdx.x >> 5;
    const int gwave  = blockIdx.x * 4 + w;
    const int b      = gwave >> 13;          // / (64*128)
    const int r      = gwave & 8191;
    const int c0     = (r >> 7) << 4;
    const int t0     = (r & 127) << 4;
    const int half16 = lane >> 4;
    const int l15    = lane & 15;

    v8f acc = zero8();
    const half_t* arow = Aw + (size_t)(c0 + l15) * 640;
    const half_t* xbase = xpad + ((size_t)b * TP + t0 + l15) * KD;

    for (int kk0 = 0; kk0 < 640; kk0 += 32) {
        HFrag a, bf;
        load_a_frag(a, arow + kk0, half16);
        const int j     = kk0 >> 7;
        const int ibase = kk0 & 127;
        load_b_frag(bf, xbase + (size_t)(2 * j) * KD + ibase, half16);
        if (kk0 + 32 < 640) __builtin_prefetch(arow + kk0 + 32, 0, 0);
        acc = __builtin_amdgcn_wmma_f32_16x16x32_f16(
            false, a.v, false, bf.v, (short)0, acc, false, false);
    }

    const int t  = t0 + l15;
    const int mb = b * 8 + (t >> 8);
    const int d  = (c0 >> 3) + half16;         // constant per lane
    const int s0 = (t & 255) << 3;             // s = s0 + v (hh == v)
#pragma unroll
    for (int v = 0; v < 8; ++v) {
        int c = c0 + v + 8 * half16;
        float val = acc[v] + bias[c];
        val = silu_f(val) * scale;
        out[((size_t)mb * 2048 + s0 + v) * 128 + d] = (half_t)val;
    }
}

// ---------------- conv V (pointwise) ----------------
// silu( sum_i x[b,t,i] * Av[c,i] ), stored transposed Vt[m][d][s] (one b128 store/lane)
__global__ void __launch_bounds__(128) conv_v_kernel(
    const half_t* __restrict__ xpad,  // [B,2056,128] f16 (row t+8 == x[b,t])
    const half_t* __restrict__ Av,    // [1024,128] f16
    half_t* __restrict__ Vt) {        // [32,128,2048] f16 (scrambled, d-major)
    const int lane   = threadIdx.x & 31;
    const int w      = threadIdx.x >> 5;
    const int gwave  = blockIdx.x * 4 + w;
    const int b      = gwave >> 13;
    const int r      = gwave & 8191;
    const int c0     = (r >> 7) << 4;
    const int t0     = (r & 127) << 4;
    const int half16 = lane >> 4;
    const int l15    = lane & 15;

    v8f acc = zero8();
    const half_t* arow = Av + (size_t)(c0 + l15) * 128;
    const half_t* xrow = xpad + ((size_t)b * TP + 8 + t0 + l15) * KD;

#pragma unroll
    for (int kk0 = 0; kk0 < 128; kk0 += 32) {
        HFrag a, bf;
        load_a_frag(a, arow + kk0, half16);
        load_b_frag(bf, xrow + kk0, half16);
        acc = __builtin_amdgcn_wmma_f32_16x16x32_f16(
            false, a.v, false, bf.v, (short)0, acc, false, false);
    }

    const int t  = t0 + l15;
    const int mb = b * 8 + (t >> 8);
    const int d  = (c0 >> 3) + half16;
    const int s0 = (t & 255) << 3;
    Pack8 pk;
#pragma unroll
    for (int v = 0; v < 8; ++v) pk.h[v] = (half_t)silu_f(acc[v]);
    *(v4u*)(Vt + ((size_t)mb * 128 + d) * 2048 + s0) = pk.q;
}

// ---------------- flash attention (causal, per scrambled batch m) ----------------
__global__ void __launch_bounds__(128) attn_kernel(
    const half_t* __restrict__ Q2,  // [32,2048,128]
    const half_t* __restrict__ K2,  // [32,2048,128]
    const half_t* __restrict__ Vt,  // [32,128,2048]
    half_t* __restrict__ Yh) {      // [4,2048,1024]  Yh[b][t][h*128+d]
    __shared__ _Float16 lds_p[4][16 * 32];  // per-wave P staging (D-layout -> A-layout)

    const int lane   = threadIdx.x & 31;
    const int w      = threadIdx.x >> 5;
    const int gwave  = blockIdx.x * 4 + w;
    const int m      = gwave >> 7;   // batch 0..31
    const int qt     = gwave & 127;  // 16-row q tile
    const int half16 = lane >> 4;
    const int l15    = lane & 15;
    const float NEG_INF = -__builtin_inff();

    // Q fragments: 16 rows x 128 d = 4 A-frags (already pre-scaled by K^-0.25)
    HFrag qf[4];
    {
        const half_t* qrow = Q2 + ((size_t)m * 2048 + qt * 16 + l15) * 128;
#pragma unroll
        for (int dc = 0; dc < 4; ++dc) load_a_frag(qf[dc], qrow + dc * 32, half16);
    }

    v8f o[8];
    float rowM[8], rowL[8];
#pragma unroll
    for (int i = 0; i < 8; ++i) { o[i] = zero8(); rowM[i] = NEG_INF; rowL[i] = 0.f; }

    _Float16* pbuf = lds_p[w];
    const int kbmax = (qt * 16 + 15) >> 5;

    for (int kb = 0; kb <= kbmax; ++kb) {
        const int sk0 = kb * 32;

        // S = Q K^T for two 16-wide key tiles
        v8f sA[2];
        sA[0] = zero8();
        sA[1] = zero8();
#pragma unroll
        for (int nn = 0; nn < 2; ++nn) {
            const half_t* krow =
                K2 + ((size_t)m * 2048 + sk0 + nn * 16 + l15) * 128;
#pragma unroll
            for (int dc = 0; dc < 4; ++dc) {
                HFrag kf;
                load_b_frag(kf, krow + dc * 32, half16);
                sA[nn] = __builtin_amdgcn_wmma_f32_16x16x32_f16(
                    false, qf[dc].v, false, kf.v, (short)0, sA[nn], false, false);
            }
        }

        // online softmax: causal mask, 16-lane row reductions, P -> LDS (f16)
        float sc[8];
        const int skA = sk0 + l15;
        const int skB = sk0 + 16 + l15;
#pragma unroll
        for (int v = 0; v < 8; ++v) {
            int sq = qt * 16 + v + 8 * half16;
            float a0 = (skA <= sq) ? sA[0][v] : NEG_INF;
            float a1 = (skB <= sq) ? sA[1][v] : NEG_INF;
            float tmx = fmaxf(a0, a1);
            tmx = fmaxf(tmx, __shfl_xor(tmx, 1, 32));
            tmx = fmaxf(tmx, __shfl_xor(tmx, 2, 32));
            tmx = fmaxf(tmx, __shfl_xor(tmx, 4, 32));
            tmx = fmaxf(tmx, __shfl_xor(tmx, 8, 32));
            float mnew = fmaxf(rowM[v], tmx);
            float so   = __expf(rowM[v] - mnew);
            float e0   = __expf(a0 - mnew);
            float e1   = __expf(a1 - mnew);
            float rs   = e0 + e1;
            rs += __shfl_xor(rs, 1, 32);
            rs += __shfl_xor(rs, 2, 32);
            rs += __shfl_xor(rs, 4, 32);
            rs += __shfl_xor(rs, 8, 32);
            rowL[v] = rowL[v] * so + rs;
            rowM[v] = mnew;
            sc[v]   = so;
            int rr = v + 8 * half16;
            pbuf[rr * 32 + l15]      = (half_t)e0;
            pbuf[rr * 32 + 16 + l15] = (half_t)e1;
        }

        // rescale accumulators
#pragma unroll
        for (int dt = 0; dt < 8; ++dt)
#pragma unroll
            for (int v = 0; v < 8; ++v) o[dt][v] *= sc[v];

        // wait for per-wave LDS writes before cross-lane reads (CDNA5 split counter)
        asm volatile("s_wait_dscnt 0" ::: "memory");

        // reload P as A-fragment (16 sq x 32 sk) -- ds_load_b128 x2
        HFrag pf;
        load_a_frag(pf, pbuf + l15 * 32, half16);

        // O += P * V
#pragma unroll
        for (int dt = 0; dt < 8; ++dt) {
            const half_t* vcol =
                Vt + ((size_t)m * 128 + dt * 16 + l15) * 2048 + sk0;
            HFrag vf;
            load_b_frag(vf, vcol, half16);
            o[dt] = __builtin_amdgcn_wmma_f32_16x16x32_f16(
                false, pf.v, false, vf.v, (short)0, o[dt], false, false);
        }
    }

    // normalize and store to projection layout Yh[b][t][h*128+d]
    const int bq = m >> 3, h2 = m & 7;
#pragma unroll
    for (int v = 0; v < 8; ++v) {
        float inv = 1.0f / rowL[v];
        int sq = qt * 16 + v + 8 * half16;
        half_t* yrow = Yh + ((size_t)bq * 2048 + sq) * 1024 + h2 * 128;
#pragma unroll
        for (int dt = 0; dt < 8; ++dt)
            yrow[dt * 16 + l15] = (half_t)(o[dt][v] * inv);
    }
}

// ---------------- output projection: silu(Y @ Wu^T + bu) ----------------
__global__ void __launch_bounds__(128) proj_kernel(
    const half_t* __restrict__ Yh,   // [4,2048,1024]
    const half_t* __restrict__ Wuh,  // [128,1024]
    const float*  __restrict__ bu,   // [128]
    float* __restrict__ out) {       // [4,2048,128]
    const int lane   = threadIdx.x & 31;
    const int w      = threadIdx.x >> 5;
    const int gwave  = blockIdx.x * 4 + w;
    const int b      = gwave >> 7;
    const int t0     = (gwave & 127) << 4;
    const int half16 = lane >> 4;
    const int l15    = lane & 15;

    v8f acc[8];
#pragma unroll
    for (int i = 0; i < 8; ++i) acc[i] = zero8();

    const half_t* yrow = Yh + ((size_t)b * 2048 + t0 + l15) * 1024;
    const half_t* wrow0 = Wuh + (size_t)l15 * 1024;

    for (int kk0 = 0; kk0 < 1024; kk0 += 32) {
        HFrag a;
        load_a_frag(a, yrow + kk0, half16);
        if (kk0 + 32 < 1024) __builtin_prefetch(yrow + kk0 + 32, 0, 0);
#pragma unroll
        for (int ct = 0; ct < 8; ++ct) {
            HFrag bf;
            load_b_frag(bf, wrow0 + (size_t)(ct * 16) * 1024 + kk0, half16);
            acc[ct] = __builtin_amdgcn_wmma_f32_16x16x32_f16(
                false, a.v, false, bf.v, (short)0, acc[ct], false, false);
        }
    }

#pragma unroll
    for (int ct = 0; ct < 8; ++ct) {
        int c = ct * 16 + l15;
        float bias = bu[c];
#pragma unroll
        for (int v = 0; v < 8; ++v) {
            int t2 = t0 + v + 8 * half16;
            float val = silu_f(acc[ct][v] + bias);
            out[((size_t)b * 2048 + t2) * 128 + c] = val;
        }
    }
}

// ---------------- launcher ----------------

extern "C" void kernel_launch(void* const* d_in, const int* in_sizes, int n_in,
                              void* d_out, int out_size, void* d_ws,
                              size_t ws_size, hipStream_t stream) {
    const float* x  = (const float*)d_in[0];
    const float* Wq = (const float*)d_in[1];
    const float* bq = (const float*)d_in[2];
    const float* Wk = (const float*)d_in[3];
    const float* bk = (const float*)d_in[4];
    const float* Wv = (const float*)d_in[5];
    const float* Wu = (const float*)d_in[6];
    const float* bu = (const float*)d_in[7];
    float* out = (float*)d_out;

    half_t* ws = (half_t*)d_ws;
    // region offsets in halfs (all multiples of 8 -> 16B aligned)
    half_t* xpad = ws;                 // 1,052,672  (4*2056*128)
    half_t* Aq   = ws + 1052672;       //   655,360
    half_t* Ak   = ws + 1708032;       //   655,360
    half_t* Av   = ws + 2363392;       //   131,072
    half_t* Wuh  = ws + 2494464;       //   131,072
    half_t* Q2   = ws + 2625536;       // 8,388,608
    half_t* K2   = ws + 11014144;      // 8,388,608
    half_t* Vt   = ws + 19402752;      // 8,388,608
    half_t* Yh   = ws + 27791360;      // 8,388,608  (total ~69 MB)

    const float qk_scale = 0.29730177875068026f;  // 128^-0.25

    // pack
    pack_xpad_kernel<<<(BB * TP * KD + 255) / 256, 256, 0, stream>>>(x, xpad);
    repack_w_kernel<<<2560, 256, 0, stream>>>(Wq, Aq);
    repack_w_kernel<<<2560, 256, 0, stream>>>(Wk, Ak);
    cvt_f32_f16_kernel<<<512, 256, 0, stream>>>(Wv, Av, 131072);
    cvt_f32_f16_kernel<<<512, 256, 0, stream>>>(Wu, Wuh, 131072);

    // convs (32768 waves each, 4 waves/block)
    conv_qk_kernel<<<8192, 128, 0, stream>>>(xpad, Aq, bq, Q2, qk_scale);
    conv_qk_kernel<<<8192, 128, 0, stream>>>(xpad, Ak, bk, K2, qk_scale);
    conv_v_kernel<<<8192, 128, 0, stream>>>(xpad, Av, Vt);

    // flash attention: 32 batches x 128 q-tiles = 4096 waves
    attn_kernel<<<1024, 128, 0, stream>>>(Q2, K2, Vt, Yh);

    // projection: 4 x 128 t-tiles = 512 waves
    proj_kernel<<<128, 128, 0, stream>>>(Yh, Wuh, bu, out);

    (void)in_sizes; (void)n_in; (void)out_size; (void)ws_size;
}
